// LTCCell_76295799046670
// MI455X (gfx1250) — matrix operand order repaired and checked by reading, not compiled, not run on hardware
//
#include <hip/hip_runtime.h>
#include <hip/hip_bf16.h>

// ---------------------------------------------------------------------------
// LTC cell on gfx1250: 1 input GEMM + 50 recurrent GEMM+gate steps.
// GEMMs run on the WMMA pipe as bf16 hi/lo split (3x v_wmma_f32_16x16x32_bf16
// per product) with f32 accumulation: ~8x MACs/instr vs f32 WMMA, near-fp32
// accuracy. Everything lives in L2 (192MB), so we are matrix-pipe bound.
// ---------------------------------------------------------------------------

typedef __bf16 bf16_t;
typedef __attribute__((ext_vector_type(16))) __bf16 v16bf;
typedef __attribute__((ext_vector_type(8)))  __bf16 v8bf;
typedef __attribute__((ext_vector_type(8)))  float  v8f;

#define B_DIM  4096
#define IN_DIM 512
#define H_DIM  1024

#define WMMA_BF16(A, Bf, C) \
  __builtin_amdgcn_wmma_f32_16x16x32_bf16(false, (A), false, (Bf), (short)0, (C), false, false)

__device__ __forceinline__ void bf16_split(float v, bf16_t& hi, bf16_t& lo) {
  hi = (bf16_t)v;
  lo = (bf16_t)(v - (float)hi);
}

// A fragment (16x32 bf16, row-major source [M][K]):
// lanes 0-15: row m0+l, K = k0+{0..7, 16..23}; lanes 16-31: K = k0+{8..15, 24..31}
__device__ __forceinline__ v16bf load_fragA(const bf16_t* __restrict__ base,
                                            int row, int ld, int k0, int aoff) {
  const bf16_t* p = base + (size_t)row * ld + (k0 + aoff);
  v8bf x0 = *(const v8bf*)(p);
  v8bf x1 = *(const v8bf*)(p + 16);
  v16bf r;
#pragma unroll
  for (int i = 0; i < 8; ++i) { r[i] = x0[i]; r[i + 8] = x1[i]; }
  return r;
}

// B fragment (32x16 bf16) loaded from transposed weights Wt[N][K]:
// lanes 0-15: col n0+l, K = k0+{0..7, 16..23}; lanes 16-31: K = k0+{8..15, 24..31}
__device__ __forceinline__ v16bf load_fragB(const bf16_t* __restrict__ base,
                                            int row, int ld, int k0, int boff) {
  const bf16_t* p = base + (size_t)row * ld + (k0 + boff);
  v8bf x0 = *(const v8bf*)(p);
  v8bf x1 = *(const v8bf*)(p + 16);
  v16bf r;
#pragma unroll
  for (int i = 0; i < 8; ++i) { r[i] = x0[i]; r[i + 8] = x1[i]; }
  return r;
}

// C/D 16x16 f32 tile layout: VGPR g, lane -> row = g + (lane>=16 ? 8 : 0), col = lane&15
template<bool STEP>
__device__ __forceinline__ void epilogue_tile(
    v8f acc, int mrow, int c,
    const float* __restrict__ E, const float* __restrict__ h_cur,
    float mu_c, float sig_c, float dec_c, float gn_c, float bias_c,
    float* __restrict__ out_f32, bf16_t* __restrict__ out_hi, bf16_t* __restrict__ out_lo)
{
#pragma unroll
  for (int g = 0; g < 8; ++g) {
    size_t idx = (size_t)(mrow + g) * H_DIM + c;
    if (STEP) {
      float pre = acc[g] + E[idx];                       // b_hh folded into E
      float z   = (pre - mu_c) * sig_c;
      float w   = 1.0f / (1.0f + __expf(-z));            // sigmoid gate
      float hn  = dec_c * h_cur[idx] + gn_c * (w * pre); // Euler update
      out_f32[idx] = hn;
      bf16_t hi, lo; bf16_split(hn, hi, lo);
      out_hi[idx] = hi; out_lo[idx] = lo;
    } else {
      out_f32[idx] = acc[g] + bias_c;                    // E = x@W_in + b_in + b_hh
    }
  }
}

// 8 waves / block; wave grid 4(M) x 2(N); each wave -> 32x32 C (2x2 WMMA tiles).
// Block tile 128x64; bf16 hi/lo split => 12 WMMAs per K=32 chunk per wave.
template<bool STEP>
__global__ __launch_bounds__(256)
void ltc_gemm(const bf16_t* __restrict__ A_hi, const bf16_t* __restrict__ A_lo, int lda,
              const bf16_t* __restrict__ Bt_hi, const bf16_t* __restrict__ Bt_lo, int ldb,
              int K,
              const float* __restrict__ E,
              const float* __restrict__ b_in, const float* __restrict__ b_hh,
              const float* __restrict__ mu, const float* __restrict__ sigma,
              const float* __restrict__ decay, const float* __restrict__ gain,
              const float* __restrict__ h_cur,
              float* __restrict__ out_f32,
              bf16_t* __restrict__ out_hi, bf16_t* __restrict__ out_lo)
{
  const int lane = threadIdx.x & 31;
  const int wave = threadIdx.x >> 5;
  const int wm   = wave & 3;
  const int wn   = wave >> 2;
  const int m0   = blockIdx.y * 128 + wm * 32;
  const int n0   = blockIdx.x * 64  + wn * 32;
  const int l    = lane & 15;
  const bool up  = lane >= 16;
  const int aoff = up ? 8  : 0;
  const int boff = up ? 8 : 0;

  const int arow0 = m0 + l,      arow1 = m0 + 16 + l;
  const int brow0 = n0 + l,      brow1 = n0 + 16 + l;

  v8f acc00 = {}, acc01 = {}, acc10 = {}, acc11 = {};

#pragma unroll 2
  for (int k0 = 0; k0 < K; k0 += 32) {
    v16bf a0h = load_fragA(A_hi, arow0, lda, k0, aoff);
    v16bf a1h = load_fragA(A_hi, arow1, lda, k0, aoff);
    v16bf a0l = load_fragA(A_lo, arow0, lda, k0, aoff);
    v16bf a1l = load_fragA(A_lo, arow1, lda, k0, aoff);
    v16bf b0h = load_fragB(Bt_hi, brow0, ldb, k0, boff);
    v16bf b1h = load_fragB(Bt_hi, brow1, ldb, k0, boff);
    v16bf b0l = load_fragB(Bt_lo, brow0, ldb, k0, boff);
    v16bf b1l = load_fragB(Bt_lo, brow1, ldb, k0, boff);

    acc00 = WMMA_BF16(a0h, b0h, acc00);
    acc01 = WMMA_BF16(a0h, b1h, acc01);
    acc10 = WMMA_BF16(a1h, b0h, acc10);
    acc11 = WMMA_BF16(a1h, b1h, acc11);
    acc00 = WMMA_BF16(a0h, b0l, acc00);
    acc01 = WMMA_BF16(a0h, b1l, acc01);
    acc10 = WMMA_BF16(a1h, b0l, acc10);
    acc11 = WMMA_BF16(a1h, b1l, acc11);
    acc00 = WMMA_BF16(a0l, b0h, acc00);
    acc01 = WMMA_BF16(a0l, b1h, acc01);
    acc10 = WMMA_BF16(a1l, b0h, acc10);
    acc11 = WMMA_BF16(a1l, b1h, acc11);
  }

  const int c0    = n0 + l;
  const int c1    = n0 + 16 + l;
  const int mrow0 = m0 + (up ? 8 : 0);
  const int mrow1 = m0 + 16 + (up ? 8 : 0);

  if (STEP) {
    float mu0 = mu[c0],    mu1 = mu[c1];
    float sg0 = sigma[c0], sg1 = sigma[c1];
    float dc0 = decay[c0], dc1 = decay[c1];
    float gn0 = gain[c0],  gn1 = gain[c1];
    epilogue_tile<true>(acc00, mrow0, c0, E, h_cur, mu0, sg0, dc0, gn0, 0.f, out_f32, out_hi, out_lo);
    epilogue_tile<true>(acc01, mrow0, c1, E, h_cur, mu1, sg1, dc1, gn1, 0.f, out_f32, out_hi, out_lo);
    epilogue_tile<true>(acc10, mrow1, c0, E, h_cur, mu0, sg0, dc0, gn0, 0.f, out_f32, out_hi, out_lo);
    epilogue_tile<true>(acc11, mrow1, c1, E, h_cur, mu1, sg1, dc1, gn1, 0.f, out_f32, out_hi, out_lo);
  } else {
    float bb0 = b_in[c0] + b_hh[c0];
    float bb1 = b_in[c1] + b_hh[c1];
    epilogue_tile<false>(acc00, mrow0, c0, nullptr, nullptr, 0, 0, 0, 0, bb0, out_f32, nullptr, nullptr);
    epilogue_tile<false>(acc01, mrow0, c1, nullptr, nullptr, 0, 0, 0, 0, bb1, out_f32, nullptr, nullptr);
    epilogue_tile<false>(acc10, mrow1, c0, nullptr, nullptr, 0, 0, 0, 0, bb0, out_f32, nullptr, nullptr);
    epilogue_tile<false>(acc11, mrow1, c1, nullptr, nullptr, 0, 0, 0, 0, bb1, out_f32, nullptr, nullptr);
  }
}

// ---- one-time prep kernels -------------------------------------------------

__global__ void prep_coeffs(const float* __restrict__ log_tau,
                            const float* __restrict__ gleak,
                            const float* __restrict__ cm_t,
                            float* __restrict__ decF, float* __restrict__ gnF,
                            float* __restrict__ decR, float* __restrict__ gnR,
                            float dt_full, float dt_rem)
{
  int i = blockIdx.x * blockDim.x + threadIdx.x;
  if (i >= H_DIM) return;
  float lt  = log_tau[i];
  float sp  = (lt > 20.0f) ? lt : log1pf(__expf(lt));  // softplus
  float tau = sp + 0.001f;
  float icg = 1.0f / (cm_t[i] + gleak[i]);
  decF[i] = 1.0f - dt_full / tau;
  gnF[i]  = dt_full * icg;
  decR[i] = 1.0f - dt_rem / tau;
  gnR[i]  = dt_rem * icg;
}

// dst[cols][rows] = hi/lo bf16 split of src[rows][cols] (one-time W transpose)
__global__ void transpose_split(const float* __restrict__ src, int rows, int cols,
                                bf16_t* __restrict__ dst_hi, bf16_t* __restrict__ dst_lo)
{
  int n = blockIdx.x * 16 + threadIdx.x;  // src col
  int k = blockIdx.y * 16 + threadIdx.y;  // src row
  if (n >= cols || k >= rows) return;
  float v = src[(size_t)k * cols + n];
  bf16_t hi, lo; bf16_split(v, hi, lo);
  dst_hi[(size_t)n * rows + k] = hi;
  dst_lo[(size_t)n * rows + k] = lo;
}

__global__ void split_f32(const float* __restrict__ src, int n,
                          bf16_t* __restrict__ hi, bf16_t* __restrict__ lo,
                          float* __restrict__ copy)
{
  int i = blockIdx.x * blockDim.x + threadIdx.x;
  if (i >= n) return;
  float v = src[i];
  bf16_t h, l; bf16_split(v, h, l);
  hi[i] = h; lo[i] = l;
  if (copy) copy[i] = v;
}

// ---------------------------------------------------------------------------

extern "C" void kernel_launch(void* const* d_in, const int* in_sizes, int n_in,
                              void* d_out, int out_size, void* d_ws, size_t ws_size,
                              hipStream_t stream)
{
  const float* x       = (const float*)d_in[0];
  const float* h_init  = (const float*)d_in[1];
  const float* W_in    = (const float*)d_in[2];
  const float* b_in    = (const float*)d_in[3];
  const float* W_hh    = (const float*)d_in[4];
  const float* b_hh    = (const float*)d_in[5];
  const float* log_tau = (const float*)d_in[6];
  const float* mu      = (const float*)d_in[7];
  const float* sigma   = (const float*)d_in[8];
  const float* gleak   = (const float*)d_in[9];
  const float* cm_t    = (const float*)d_in[10];
  // d_in[11] = dt (python scalar, == 1): schedule is static, mirroring the
  // reference which derives loop counts from the python float at trace time.

  char* p = (char*)d_ws;
  size_t off = 0;
  auto carve = [&](size_t bytes) -> char* {
    char* q = p + off;
    off += (bytes + 255) & ~(size_t)255;
    return q;
  };

  const size_t BH  = (size_t)B_DIM * H_DIM;
  const size_t BI  = (size_t)B_DIM * IN_DIM;
  const size_t HH  = (size_t)H_DIM * H_DIM;
  const size_t HI  = (size_t)H_DIM * IN_DIM;

  float*  E       = (float*)carve(BH * 4);          // input_effect + b_in + b_hh
  float*  hF[2]   = {(float*)carve(BH * 4), (float*)carve(BH * 4)};
  bf16_t* hHi[2]  = {(bf16_t*)carve(BH * 2), (bf16_t*)carve(BH * 2)};
  bf16_t* hLo[2]  = {(bf16_t*)carve(BH * 2), (bf16_t*)carve(BH * 2)};
  bf16_t* xHi     = (bf16_t*)carve(BI * 2);
  bf16_t* xLo     = (bf16_t*)carve(BI * 2);
  bf16_t* WhhT_hi = (bf16_t*)carve(HH * 2);
  bf16_t* WhhT_lo = (bf16_t*)carve(HH * 2);
  bf16_t* WinT_hi = (bf16_t*)carve(HI * 2);
  bf16_t* WinT_lo = (bf16_t*)carve(HI * 2);
  float*  decF    = (float*)carve(H_DIM * 4);
  float*  gnF     = (float*)carve(H_DIM * 4);
  float*  decR    = (float*)carve(H_DIM * 4);
  float*  gnR     = (float*)carve(H_DIM * 4);

  // Step schedule (mirrors reference double-precision arithmetic, dt = 1):
  // n_full = floor(1.0/0.02) = 49, rem = 1 - 49*0.02 ~= 0.02 > 1e-8 -> 50 steps
  const double MAXDT = 0.02;
  double dt_val = 1.0;
  int n_full = (int)(dt_val / MAXDT);
  if ((double)n_full * MAXDT > dt_val) n_full -= 1;   // floor-division guard
  double rem = dt_val - (double)n_full * MAXDT;
  bool has_rem = rem > 1e-8;
  int n_steps = n_full + (has_rem ? 1 : 0);

  // --- one-time prep ---
  prep_coeffs<<<(H_DIM + 255) / 256, 256, 0, stream>>>(
      log_tau, gleak, cm_t, decF, gnF, decR, gnR, (float)MAXDT, (float)rem);

  dim3 tb(16, 16);
  transpose_split<<<dim3(H_DIM / 16, IN_DIM / 16), tb, 0, stream>>>(
      W_in, IN_DIM, H_DIM, WinT_hi, WinT_lo);
  transpose_split<<<dim3(H_DIM / 16, H_DIM / 16), tb, 0, stream>>>(
      W_hh, H_DIM, H_DIM, WhhT_hi, WhhT_lo);

  split_f32<<<(int)((BI + 255) / 256), 256, 0, stream>>>(x, (int)BI, xHi, xLo, nullptr);
  split_f32<<<(int)((BH + 255) / 256), 256, 0, stream>>>(h_init, (int)BH, hHi[0], hLo[0], hF[0]);

  // --- input GEMM: E = x @ W_in + b_in + b_hh ---
  dim3 ggrid(H_DIM / 64, B_DIM / 128);
  ltc_gemm<false><<<ggrid, 256, 0, stream>>>(
      xHi, xLo, IN_DIM, WinT_hi, WinT_lo, IN_DIM, IN_DIM,
      nullptr, b_in, b_hh, nullptr, nullptr, nullptr, nullptr, nullptr,
      E, nullptr, nullptr);

  if (n_steps == 0) {
    hipMemcpyAsync(d_out, (const void*)h_init, BH * sizeof(float),
                   hipMemcpyDeviceToDevice, stream);
    return;
  }

  // --- 50 sequential fused GEMM + liquid-gate Euler steps ---
  int cur = 0;
  for (int s = 0; s < n_steps; ++s) {
    int nxt = cur ^ 1;
    bool last = (s == n_steps - 1);
    bool full = (s < n_full);
    ltc_gemm<true><<<ggrid, 256, 0, stream>>>(
        hHi[cur], hLo[cur], H_DIM, WhhT_hi, WhhT_lo, H_DIM, H_DIM,
        E, nullptr, nullptr, mu, sigma,
        full ? decF : decR, full ? gnF : gnR,
        hF[cur],
        last ? (float*)d_out : hF[nxt], hHi[nxt], hLo[nxt]);
    cur = nxt;
  }
}
